// RNNAttention_22273700397550
// MI455X (gfx1250) — matrix operand order repaired and checked by reading, not compiled, not run on hardware
//
#include <hip/hip_runtime.h>
#include <stdint.h>

#define VOCAB 32000
#define H 256
#define B 4
#define S 512
#define MROWS (B*S)   // 2048

typedef __attribute__((ext_vector_type(16))) __bf16 v16bf;
typedef __attribute__((ext_vector_type(8)))  float  v8f;
typedef int vi4 __attribute__((vector_size(16)));   // matches builtin param type

union FragBF { uint4 u[2]; v16bf v; };

#if __has_builtin(__builtin_amdgcn_global_load_async_to_lds_b128)
#define ASYNC_LDS 1
#else
#define ASYNC_LDS 0
#endif

// 16-byte global -> LDS stage: async DMA (ASYNCcnt) on gfx1250, else direct copy
__device__ __forceinline__ void stage16(const uint4* gsrc, unsigned int* ldst) {
#if ASYNC_LDS
  __builtin_amdgcn_global_load_async_to_lds_b128(
      (__attribute__((address_space(1))) vi4*)(unsigned long long)gsrc,
      (__attribute__((address_space(3))) vi4*)ldst,
      0, 0);
#else
  *(uint4*)ldst = *gsrc;
#endif
}

__device__ __forceinline__ void waitAsync() {
#if ASYNC_LDS
#if __has_builtin(__builtin_amdgcn_s_wait_asynccnt)
  __builtin_amdgcn_s_wait_asynccnt(0);
#else
  asm volatile("s_wait_asynccnt 0x0" ::: "memory");
#endif
#endif
}

// hardware transcendental tanh (v_tanh_f32)
__device__ __forceinline__ float fast_tanh(float x) {
#if __has_builtin(__builtin_amdgcn_tanhf)
  return __builtin_amdgcn_tanhf(x);
#else
  float r;
  asm("v_tanh_f32 %0, %1" : "=v"(r) : "v"(x));
  return r;
#endif
}

__device__ __forceinline__ unsigned short f2bf_rn(float f) {
  unsigned int u = __float_as_uint(f);
  unsigned int r = u + 0x7FFFu + ((u >> 16) & 1u);
  return (unsigned short)(r >> 16);
}
__device__ __forceinline__ float bf2f(unsigned short h) {
  return __uint_as_float(((unsigned int)h) << 16);
}

// -------- K1: fused embedding gather + pre = emb@Wi + bi + bh --------------
__global__ void __launch_bounds__(256) k_embed_pre(
    const int* __restrict__ x, const float* __restrict__ embed,
    const float* __restrict__ Wi, const float* __restrict__ bi,
    const float* __restrict__ bh, float* __restrict__ pre) {
  int m = blockIdx.x;          // 0..2047 (b*S+t)
  int j = threadIdx.x;         // 0..255
  __shared__ float er[H];
  int row = x[m];
  er[j] = embed[row * H + j];
  __syncthreads();
  float acc = bi[j] + bh[j];
#pragma unroll 8
  for (int k = 0; k < H; ++k) acc += er[k] * Wi[k * H + j];
  pre[m * H + j] = acc;
}

// -------- K2: sequential RNN scan, Wh column resident in VGPRs -------------
__global__ void __launch_bounds__(256) k_rnn(
    const float* __restrict__ pre, const float* __restrict__ Wh,
    float* __restrict__ enc, float* __restrict__ outTail) {
  int b = blockIdx.x;
  int j = threadIdx.x;
  float4 w[64];                          // Wh[:, j] -> 256 VGPRs (wave32 has 1024)
#pragma unroll
  for (int k4 = 0; k4 < 64; ++k4) {
    w[k4].x = Wh[(4 * k4 + 0) * H + j];
    w[k4].y = Wh[(4 * k4 + 1) * H + j];
    w[k4].z = Wh[(4 * k4 + 2) * H + j];
    w[k4].w = Wh[(4 * k4 + 3) * H + j];
  }
  __shared__ __align__(16) float hl[H];
  hl[j] = 0.f;
  __syncthreads();
  const float4* h4 = (const float4*)hl;  // broadcast ds_load_b128, conflict-free
  const float* prow = pre + (size_t)b * S * H;
  float* erow = enc + (size_t)b * S * H;
  for (int t = 0; t < S; ++t) {
    float acc = prow[t * H + j];
#pragma unroll
    for (int k4 = 0; k4 < 64; ++k4) {
      float4 hv = h4[k4];
      acc += hv.x * w[k4].x + hv.y * w[k4].y + hv.z * w[k4].z + hv.w * w[k4].w;
    }
    float hj = tanhf(acc);               // accurate: 512-step feedback loop
    __syncthreads();                     // all reads of hl done
    hl[j] = hj;
    erow[t * H + j] = hj;
    __syncthreads();                     // new h visible
  }
  outTail[b * H + j] = hl[j];            // enc[:, -1, :]
}

// -------- K3: q = enc@Wq ; kT[h][m] = (enc@Wk)^T ---------------------------
__global__ void __launch_bounds__(256) k_qk(
    const float* __restrict__ enc, const float* __restrict__ Wa,
    float* __restrict__ q, float* __restrict__ kT) {
  int m = blockIdx.x, j = threadIdx.x;
  __shared__ float er[H];
  er[j] = enc[m * H + j];
  __syncthreads();
  float aq = 0.f, ak = 0.f;
#pragma unroll 4
  for (int k = 0; k < H; ++k) {
    float e = er[k];
    aq += e * Wa[k * H + j];
    ak += e * Wa[(H + k) * H + j];
  }
  q[m * H + j] = aq;
  kT[j * MROWS + m] = ak;               // transposed for coalesced score reads
}

// -------- K4: additive-attention scores + causal softmax + ctx -------------
__global__ void __launch_bounds__(256) k_attn(
    const float* __restrict__ q, const float* __restrict__ kT,
    const float* __restrict__ enc, const float* __restrict__ va,
    const float* __restrict__ ba, float* __restrict__ ctxenc) {
  int bb = blockIdx.x / S;
  int t  = blockIdx.x % S;
  int j  = threadIdx.x;
  int mbase = bb * S;
  if (t == 0) {                          // ctx[t=0] = enc[:,0,:]  -> ctx+enc = 2*enc
    float e = enc[mbase * H + j];
    ctxenc[mbase * H + j] = 2.f * e;
    return;
  }
  __shared__ float qb[H], vas[H], sc[2 * 256], red[256];
  qb[j]  = q[(mbase + t) * H + j] + ba[j];
  vas[j] = va[j];
  __syncthreads();
  int s0 = j, s1 = j + 256;
  bool v0 = s0 < t, v1 = s1 < t;
  float a0 = 0.f, a1 = 0.f;
  if (v0 | v1) {
    const float* kp = kT + mbase;
    for (int h = 0; h < H; ++h) {
      float qh = qb[h], vh = vas[h];
      const float* kr = kp + h * MROWS;
      if (v0) a0 += vh * fast_tanh(qh + kr[s0]);   // v_tanh_f32 trans op
      if (v1) a1 += vh * fast_tanh(qh + kr[s1]);
    }
  }
  // max
  red[j] = fmaxf(v0 ? a0 : -3.0e38f, v1 ? a1 : -3.0e38f);
  __syncthreads();
  for (int st = 128; st > 0; st >>= 1) {
    if (j < st) red[j] = fmaxf(red[j], red[j + st]);
    __syncthreads();
  }
  float mx = red[0];
  __syncthreads();
  // exp + sum
  float e0 = v0 ? __expf(a0 - mx) : 0.f;           // v_exp_f32
  float e1 = v1 ? __expf(a1 - mx) : 0.f;
  sc[s0] = e0; sc[s1] = e1;
  red[j] = e0 + e1;
  __syncthreads();
  for (int st = 128; st > 0; st >>= 1) {
    if (j < st) red[j] += red[j + st];
    __syncthreads();
  }
  float inv = 1.f / red[0];
  // ctx: j is now the channel index, coalesced enc reads
  float acc = 0.f;
  for (int s = 0; s < t; ++s) acc += sc[s] * enc[(mbase + s) * H + j];
  float et = enc[(mbase + t) * H + j];
  ctxenc[(mbase + t) * H + j] = acc * inv + et;
}

// -------- K5a: split (ctx+enc) into bf16 hi/lo -----------------------------
__global__ void __launch_bounds__(256) k_convA(
    const float* __restrict__ src, unsigned short* __restrict__ hi,
    unsigned short* __restrict__ lo) {
  int i = blockIdx.x * 256 + threadIdx.x;
  float xv = src[i];
  unsigned short h = f2bf_rn(xv);
  hi[i] = h;
  lo[i] = f2bf_rn(xv - bf2f(h));
}

// -------- K5b: transpose Wo -> WoT[n][k] and split into bf16 hi/lo ---------
__global__ void __launch_bounds__(256) k_convW(
    const float* __restrict__ Wo, unsigned short* __restrict__ Thi,
    unsigned short* __restrict__ Tlo) {
  __shared__ float tile[32][33];
  int n0 = blockIdx.x * 32;              // vocab dim
  int k0 = blockIdx.y * 32;              // H dim
  int tx = threadIdx.x & 31, ty = threadIdx.x >> 5;
#pragma unroll
  for (int i = 0; i < 4; ++i) {
    int kk = ty + 8 * i;
    tile[kk][tx] = Wo[(k0 + kk) * VOCAB + n0 + tx];
  }
  __syncthreads();
#pragma unroll
  for (int i = 0; i < 4; ++i) {
    int nn = ty + 8 * i;
    float xv = tile[tx][nn];
    unsigned short hbits = f2bf_rn(xv);
    int o = (n0 + nn) * H + k0 + tx;     // coalesced over tx
    Thi[o] = hbits;
    Tlo[o] = f2bf_rn(xv - bf2f(hbits));
  }
}

// -------- K6: out = (ctx+enc)@Wo + bo via split-bf16 WMMA ------------------
// Block tile 128(M) x 64(N), K=256 in 8 chunks of 32; 8 waves, each wave owns
// a 32x32 output (2x2 wmma tiles). Double-buffered LDS filled by
// GLOBAL_LOAD_ASYNC_TO_LDS_B128 (ASYNCcnt) so the DMA of chunk k+1 overlaps
// the 12 WMMAs of chunk k. LDS rows padded to 20 dwords (80B) so the 16B
// fragment ds_loads are bank-conflict-free.
__global__ void __launch_bounds__(256) k_gemm_out(
    const unsigned short* __restrict__ Ahi, const unsigned short* __restrict__ Alo,
    const unsigned short* __restrict__ Bhi, const unsigned short* __restrict__ Blo,
    const float* __restrict__ bo, float* __restrict__ out) {
  const int tid = threadIdx.x;
  const int n0 = blockIdx.x * 64;
  const int m0 = blockIdx.y * 128;
  __shared__ __align__(16) unsigned int sAhi[2][128 * 20];
  __shared__ __align__(16) unsigned int sAlo[2][128 * 20];
  __shared__ __align__(16) unsigned int sBhi[2][64 * 20];
  __shared__ __align__(16) unsigned int sBlo[2][64 * 20];
  const uint4* A4h = (const uint4*)Ahi;  // 32 uint4 per 256-elem bf16 row
  const uint4* A4l = (const uint4*)Alo;
  const uint4* B4h = (const uint4*)Bhi;
  const uint4* B4l = (const uint4*)Blo;

  const int wid  = tid >> 5;
  const int wm   = wid & 3;              // M sub-block (32 rows)
  const int wn   = wid >> 2;             // N sub-block (32 cols)
  const int lane = tid & 31;
  const int half = lane >> 4;
  const int lr   = lane & 15;

  // per-thread staging geometry (row/col within the block tile)
  const int rowA = tid >> 2;             // 0..63  (and +64 for second unit)
  const int c4   = tid & 3;              // which uint4 of the 32-bf16 chunk row

  auto stage = [&](int kc, int bufi) {
    int g0 = (m0 + rowA) * 32 + kc * 4 + c4;
    int s0o = rowA * 20 + c4 * 4;
    stage16(A4h + g0, &sAhi[bufi][s0o]);
    stage16(A4l + g0, &sAlo[bufi][s0o]);
    int g1 = (m0 + rowA + 64) * 32 + kc * 4 + c4;
    int s1o = (rowA + 64) * 20 + c4 * 4;
    stage16(A4h + g1, &sAhi[bufi][s1o]);
    stage16(A4l + g1, &sAlo[bufi][s1o]);
    int gb = (n0 + rowA) * 32 + kc * 4 + c4;
    stage16(B4h + gb, &sBhi[bufi][s0o]);
    stage16(B4l + gb, &sBlo[bufi][s0o]);
  };

  v8f acc[2][2];
#pragma unroll
  for (int mi = 0; mi < 2; ++mi)
#pragma unroll
    for (int ni = 0; ni < 2; ++ni)
#pragma unroll
      for (int e = 0; e < 8; ++e) acc[mi][ni][e] = 0.f;

  stage(0, 0);
  waitAsync();
  __syncthreads();

#pragma unroll
  for (int kc = 0; kc < 8; ++kc) {
    const int cur = kc & 1;
    if (kc < 7) stage(kc + 1, cur ^ 1);  // DMA next chunk while computing

    // A 16x32 bf16 frag: lane<16 -> K {0..7,16..23}; lane>=16 -> K {8..15,24..31}
    FragBF ah[2], al[2], bhf[2], blf[2];
#pragma unroll
    for (int mi = 0; mi < 2; ++mi) {
      int row = wm * 32 + mi * 16 + lr;
      int base = row * 20 + half * 4;
      ah[mi].u[0] = *(const uint4*)&sAhi[cur][base];
      ah[mi].u[1] = *(const uint4*)&sAhi[cur][base + 8];
      al[mi].u[0] = *(const uint4*)&sAlo[cur][base];
      al[mi].u[1] = *(const uint4*)&sAlo[cur][base + 8];
    }
    // B 32x16 bf16 frag: lane<16 -> K {0..7,16..23}; lane>=16 -> K {8..15,24..31}
#pragma unroll
    for (int ni = 0; ni < 2; ++ni) {
      int row = wn * 32 + ni * 16 + lr;
      int base = row * 20 + half * 4;
      bhf[ni].u[0] = *(const uint4*)&sBhi[cur][base];
      bhf[ni].u[1] = *(const uint4*)&sBhi[cur][base + 8];
      blf[ni].u[0] = *(const uint4*)&sBlo[cur][base];
      blf[ni].u[1] = *(const uint4*)&sBlo[cur][base + 8];
    }
    // split product: lo*hi + hi*lo + hi*hi  (error ~ bf16^2 ~ 1e-5)
#pragma unroll
    for (int mi = 0; mi < 2; ++mi)
#pragma unroll
      for (int ni = 0; ni < 2; ++ni) {
        acc[mi][ni] = __builtin_amdgcn_wmma_f32_16x16x32_bf16(
            false, al[mi].v, false, bhf[ni].v, (short)0, acc[mi][ni], false, false);
        acc[mi][ni] = __builtin_amdgcn_wmma_f32_16x16x32_bf16(
            false, ah[mi].v, false, blf[ni].v, (short)0, acc[mi][ni], false, false);
        acc[mi][ni] = __builtin_amdgcn_wmma_f32_16x16x32_bf16(
            false, ah[mi].v, false, bhf[ni].v, (short)0, acc[mi][ni], false, false);
      }

    if (kc < 7) {
      waitAsync();                       // chunk kc+1 landed in LDS
      __syncthreads();
    }
  }

  // epilogue: C 16x16 f32 layout: lane = N col (mod 16), VGPR v = M row (+8 hi half)
#pragma unroll
  for (int mi = 0; mi < 2; ++mi)
#pragma unroll
    for (int ni = 0; ni < 2; ++ni) {
      int col = n0 + wn * 32 + ni * 16 + lr;
      float bias = bo[col];
      int rowBase = m0 + wm * 32 + mi * 16 + half * 8;
#pragma unroll
      for (int v = 0; v < 8; ++v)
        out[(rowBase + v) * VOCAB + col] = acc[mi][ni][v] + bias;
    }
}

// ---------------------------------------------------------------------------
extern "C" void kernel_launch(void* const* d_in, const int* in_sizes, int n_in,
                              void* d_out, int out_size, void* d_ws, size_t ws_size,
                              hipStream_t stream) {
  (void)in_sizes; (void)n_in; (void)out_size; (void)ws_size;
  const int*   x     = (const int*)d_in[0];
  const float* embed = (const float*)d_in[1];
  const float* Wi    = (const float*)d_in[2];
  const float* bi    = (const float*)d_in[3];
  const float* Wh    = (const float*)d_in[4];
  const float* bh    = (const float*)d_in[5];
  const float* Wa    = (const float*)d_in[6];
  const float* ba    = (const float*)d_in[7];
  const float* va    = (const float*)d_in[8];
  const float* Wo    = (const float*)d_in[9];
  const float* bo    = (const float*)d_in[10];
  float* out = (float*)d_out;

  char* ws = (char*)d_ws;
  const size_t MB = 1024 * 1024;
  float* pre    = (float*)(ws + 0 * MB);
  float* enc    = (float*)(ws + 2 * MB);
  float* q      = (float*)(ws + 4 * MB);
  float* kT     = (float*)(ws + 6 * MB);
  float* ctxenc = (float*)(ws + 8 * MB);
  unsigned short* Ahi  = (unsigned short*)(ws + 10 * MB);
  unsigned short* Alo  = (unsigned short*)(ws + 11 * MB);
  unsigned short* WThi = (unsigned short*)(ws + 12 * MB);
  unsigned short* WTlo = (unsigned short*)(ws + 12 * MB + (size_t)VOCAB * H * 2);
  float* outTail = out + (size_t)B * S * VOCAB;

  k_convW<<<dim3(VOCAB / 32, H / 32), 256, 0, stream>>>(Wo, WThi, WTlo);
  k_embed_pre<<<MROWS, 256, 0, stream>>>(x, embed, Wi, bi, bh, pre);
  k_rnn<<<B, 256, 0, stream>>>(pre, Wh, enc, outTail);
  k_qk<<<MROWS, 256, 0, stream>>>(enc, Wa, q, kT);
  k_attn<<<MROWS, 256, 0, stream>>>(q, kT, enc, va, ba, ctxenc);
  k_convA<<<MROWS, 256, 0, stream>>>(ctxenc, Ahi, Alo);
  k_gemm_out<<<dim3(VOCAB / 64, MROWS / 128), 256, 0, stream>>>(Ahi, Alo, WThi, WTlo, bo, out);
}
